// Attention_4415226380332
// MI455X (gfx1250) — matrix operand order repaired and checked by reading, not compiled, not run on hardware
//
#include <hip/hip_runtime.h>

#define BATCH 16
#define DDIM  512
#define NDIM  1024
#define TDIM  2048
#define QKSCALE 0.04419417382415922f  // 1/sqrt(512)

typedef __attribute__((ext_vector_type(16))) __bf16 v16bf;
typedef __attribute__((ext_vector_type(8)))  float  v8f;

__device__ __forceinline__ unsigned short bfbits(float f) {
  unsigned u; __builtin_memcpy(&u, &f, 4);
  unsigned r = u + 0x7FFFu + ((u >> 16) & 1u);   // round-to-nearest-even
  return (unsigned short)(r >> 16);
}
__device__ __forceinline__ __bf16 f2bf(float f) {
  unsigned short h = bfbits(f);
  __bf16 o; __builtin_memcpy(&o, &h, 2);
  return o;
}

// ISA 7.12.2: 16-bit A-matrix 16x32 K index for (vgpr v, half h, lane-half hi)
__device__ __forceinline__ int a_kmap(int v, int h, int hi) {
  int base = (v < 4) ? ((hi ? 8 : 0) + 2 * v) : ((hi ? 24 : 16) + 2 * (v - 4));
  return base + h;
}
// k order this code uses for A and B (contiguous): lane-half hi holds K = 16*hi + j, fragment elem j = k & 15
__device__ __forceinline__ int b_kmap(int v, int h, int hi) {
  return 16 * hi + 2 * v + h;
}

// ---------------------------------------------------------------------------
// Kernel 1: A[:, :, t>=ct] = softmax_N(K^T Q * scale); A[:, :, t<ct] = prevA
// grid = (T/64, B), block = 1024 (32 waves).
// Wave w owns rows [32w, 32w+32) (2 M-tiles) x 64 columns (4 T-tiles).
// Q chunk converted to bf16 once and staged in LDS in B-fragment layout.
// ---------------------------------------------------------------------------
__global__ __launch_bounds__(1024) void attn_scores_softmax(
    const float* __restrict__ Km, const float* __restrict__ Qm,
    const float* __restrict__ prevA, const int* __restrict__ force_p,
    const int* __restrict__ ct_p, float* __restrict__ A)
{
  const int b   = blockIdx.y;
  const int t0  = blockIdx.x * 64;
  const int tid = threadIdx.x;
  const int copyLimit = (*force_p) ? (*ct_p) : 0;

  if (t0 + 64 <= copyLimit) {
    // committed prefix: pure vectorized copy of previous attention
    const int base = b * NDIM * TDIM + t0;
#pragma unroll 4
    for (int i = 0; i < 16; ++i) {
      int e = tid + i * 1024;            // 16384 float4 tiles total
      int n = e >> 4, c = (e & 15) * 4;
      int idx = base + n * TDIM + c;
      *(float4*)(A + idx) = *(const float4*)(prevA + idx);
    }
    return;
  }

  const int w    = tid >> 5;
  const int lane = tid & 31;
  const int hi   = lane >> 4;
  const int lc   = lane & 15;

  // B-fragment records: [tt][lane][j] bf16, one 32B record per (tt,lane)
  __shared__ __align__(32) unsigned short q_frag[4 * 32 * 16];   // 4 KB
  __shared__ float sm_part[4][1024];                             // 16 KB
  __shared__ float sm_col[64];

  const v8f vz = {0.f,0.f,0.f,0.f,0.f,0.f,0.f,0.f};
  v8f acc[2][4] = {{vz, vz, vz, vz}, {vz, vz, vz, vz}};

  const float* Kb = Km + b * DDIM * NDIM;
  const float* Qb = Qm + b * DDIM * TDIM;

  const int n0w = w * 32;              // this wave's row base
  const int kp  = tid >> 5;            // staging: k' within chunk (0..31)
  const int c0  = (tid & 31) * 2;      // staging: column pair (0..62)

  for (int kk = 0; kk < DDIM; kk += 32) {
    __syncthreads();   // protect q_frag reuse across iterations
    {
      // stage+convert one 32x64 Q chunk, each element converted exactly once
      const float2 q2 = *(const float2*)(Qb + (kk + kp) * TDIM + t0 + c0);
      const int hiK = kp >> 4, j = kp & 15;
      const int tt  = c0 >> 4;
      const int l0  = hiK * 16 + (c0 & 15);
      q_frag[(tt * 32 + l0) * 16 + j]     = bfbits(q2.x);
      q_frag[(tt * 32 + l0 + 1) * 16 + j] = bfbits(q2.y);
    }
    if (kk + 32 < DDIM)  // hint next K chunk toward the caches
      __builtin_prefetch(Kb + (kk + 32 + hi * 16) * NDIM + n0w + lc, 0, 1);
    __syncthreads();

    // A fragments from K^T (each element used by exactly one wave)
    v16bf af[2];
#pragma unroll
    for (int mt = 0; mt < 2; ++mt)
#pragma unroll
      for (int v = 0; v < 8; ++v)
#pragma unroll
        for (int h = 0; h < 2; ++h) {
          int k = kk + b_kmap(v, h, hi);
          af[mt][2 * v + h] = f2bf(Kb[k * NDIM + n0w + mt * 16 + lc]);
        }
#pragma unroll
    for (int tt = 0; tt < 4; ++tt) {
      const v16bf bf = *(const v16bf*)&q_frag[(tt * 32 + lane) * 16];
      acc[0][tt] = __builtin_amdgcn_wmma_f32_16x16x32_bf16(
          false, af[0], false, bf, (short)0, acc[0][tt], false, false);
      acc[1][tt] = __builtin_amdgcn_wmma_f32_16x16x32_bf16(
          false, af[1], false, bf, (short)0, acc[1][tt], false, false);
    }
  }

  // ---- softmax over N (cross-wave via LDS) ----
#pragma unroll
  for (int tt = 0; tt < 4; ++tt) {
    float lm = -3.0e38f;
#pragma unroll
    for (int mt = 0; mt < 2; ++mt)
#pragma unroll
      for (int r = 0; r < 8; ++r) {
        float s = acc[mt][tt][r] * QKSCALE;
        acc[mt][tt][r] = s;
        lm = fmaxf(lm, s);
      }
    sm_part[tt][w * 32 + lane] = lm;
  }
  __syncthreads();
  if (tid < 64) {
    const int tt = tid >> 4, col = tid & 15;
    float m = -3.0e38f;
    for (int ww = 0; ww < 32; ++ww) {
      m = fmaxf(m, sm_part[tt][ww * 32 + col]);
      m = fmaxf(m, sm_part[tt][ww * 32 + col + 16]);
    }
    sm_col[tid] = m;
  }
  __syncthreads();
#pragma unroll
  for (int tt = 0; tt < 4; ++tt) {
    const float cmax = sm_col[tt * 16 + lc];
    float ls = 0.f;
#pragma unroll
    for (int mt = 0; mt < 2; ++mt)
#pragma unroll
      for (int r = 0; r < 8; ++r) {
        float e = __expf(acc[mt][tt][r] - cmax);
        acc[mt][tt][r] = e;
        ls += e;
      }
    sm_part[tt][w * 32 + lane] = ls;
  }
  __syncthreads();
  if (tid < 64) {
    const int tt = tid >> 4, col = tid & 15;
    float s = 0.f;
    for (int ww = 0; ww < 32; ++ww)
      s += sm_part[tt][ww * 32 + col] + sm_part[tt][ww * 32 + col + 16];
    sm_col[tid] = s;
  }
  __syncthreads();

  const int baseA = b * NDIM * TDIM;
#pragma unroll
  for (int tt = 0; tt < 4; ++tt) {
    const float inv = 1.0f / sm_col[tt * 16 + lc];
    const int t = t0 + tt * 16 + lc;
    const bool usePrev = (t < copyLimit);   // straddling tile (general ct)
#pragma unroll
    for (int mt = 0; mt < 2; ++mt)
#pragma unroll
      for (int r = 0; r < 8; ++r) {
        int n = n0w + mt * 16 + hi * 8 + r;
        int idx = baseA + n * TDIM + t;
        A[idx] = usePrev ? prevA[idx] : acc[mt][tt][r] * inv;
      }
  }
}

// ---------------------------------------------------------------------------
// Kernel 2: argmax of column ct, force test, optional one-hot rewrite,
// write positions (as float, per output dtype). grid = B, block = 256.
// ---------------------------------------------------------------------------
__global__ __launch_bounds__(256) void attn_finalize(
    const int* __restrict__ prev_pos, const int* __restrict__ force_p,
    const int* __restrict__ ct_p, float* __restrict__ A,
    float* __restrict__ pos_out)
{
  const int b   = blockIdx.x;
  const int tid = threadIdx.x;
  const int ct  = *ct_p;
  __shared__ float smax[256];
  __shared__ int   sidx[256];
  __shared__ int   s_need, s_idx;

  const int base = b * NDIM * TDIM + ct;
  float best = -3.0e38f; int bidx = NDIM;
  for (int i = 0; i < 4; ++i) {
    int n = tid * 4 + i;
    float v = A[base + n * TDIM];
    if (v > best || (v == best && n < bidx)) { best = v; bidx = n; }
  }
  smax[tid] = best; sidx[tid] = bidx;
  __syncthreads();
  for (int s = 128; s > 0; s >>= 1) {
    if (tid < s) {
      float v = smax[tid + s]; int ix = sidx[tid + s];
      if (v > smax[tid] || (v == smax[tid] && ix < sidx[tid])) {
        smax[tid] = v; sidx[tid] = ix;
      }
    }
    __syncthreads();
  }
  if (tid == 0) {
    int pos = sidx[0];
    int need = 0, idx = 0;
    if (*force_p) {
      int prev = prev_pos[b];
      int diff = pos - prev;
      if (diff < -1 || diff > 3) {
        need = 1;
        idx = prev + 1;
        if (idx > NDIM - 1) idx = NDIM - 1;
        if (idx < 0) idx = 0;
        pos = idx;
      }
    }
    s_need = need; s_idx = idx;
    pos_out[b] = (float)pos;
  }
  __syncthreads();
  if (s_need) {
    for (int i = 0; i < 4; ++i) {
      int n = tid * 4 + i;
      A[base + n * TDIM] = (n == s_idx) ? 1.0f : 0.0f;
    }
  }
}

// ---------------------------------------------------------------------------
// Kernel 3: R[b,d,t] = sum_n V[b,d,n] * A[b,n,t]
// grid = (T/32, D/256, B), block = 256 (8 waves); 2x2 16x16 tiles per wave.
// All waves in a block share t0, so the attention B-fragments are converted
// once per block and staged in LDS in fragment layout (like Q in kernel 1).
// ---------------------------------------------------------------------------
__global__ __launch_bounds__(256) void r_gemm(
    const float* __restrict__ Vm, const float* __restrict__ A,
    float* __restrict__ R)
{
  const int b   = blockIdx.z;
  const int t0  = blockIdx.x * 32;
  const int tid = threadIdx.x;
  const int w    = tid >> 5;
  const int lane = tid & 31;
  const int hi   = lane >> 4;
  const int lc   = lane & 15;
  const int m0   = (blockIdx.y * 8 + w) * 32;

  const float* Vb = Vm + b * DDIM * NDIM;
  const float* Ab = A + b * NDIM * TDIM + t0;

  // B-fragment records for the 32(k)x32(t) attention chunk: [tt][lane][j]
  __shared__ __align__(32) unsigned short a_frag[2 * 32 * 16];   // 2 KB

  const v8f vz = {0.f,0.f,0.f,0.f,0.f,0.f,0.f,0.f};
  v8f acc[2][2] = {{vz, vz}, {vz, vz}};

  const int kp = tid >> 3;             // staging: k' within chunk (0..31)
  const int c0 = (tid & 7) * 4;        // staging: column quad (0..28)

  for (int k0 = 0; k0 < NDIM; k0 += 32) {
    __syncthreads();   // protect a_frag reuse across iterations
    {
      // stage+convert one 32x32 attention chunk, converted once per block
      const float4 p = *(const float4*)(Ab + (k0 + kp) * TDIM + c0);
      const int hiK = kp >> 4, j = kp & 15;
      const float pv[4] = {p.x, p.y, p.z, p.w};
#pragma unroll
      for (int i = 0; i < 4; ++i) {
        const int c = c0 + i;
        a_frag[(((c >> 4) * 32) + hiK * 16 + (c & 15)) * 16 + j] = bfbits(pv[i]);
      }
    }
    if (k0 + 32 < NDIM)  // hint next attention chunk toward the caches
      __builtin_prefetch(Ab + (k0 + 32 + kp) * TDIM + c0, 0, 1);
    __syncthreads();

    // A fragments from V (each element used by exactly one wave)
    v16bf af[2];
#pragma unroll
    for (int mt = 0; mt < 2; ++mt)
#pragma unroll
      for (int v = 0; v < 8; ++v) {
        const int kb = b_kmap(v, 0, hi);            // K pairs memory-adjacent
        const float2 p =
            *(const float2*)(Vb + (m0 + mt * 16 + lc) * NDIM + k0 + kb);
        af[mt][2 * v]     = f2bf(p.x);
        af[mt][2 * v + 1] = f2bf(p.y);
      }
#pragma unroll
    for (int tt = 0; tt < 2; ++tt) {
      const v16bf bf = *(const v16bf*)&a_frag[(tt * 32 + lane) * 16];
      acc[0][tt] = __builtin_amdgcn_wmma_f32_16x16x32_bf16(
          false, af[0], false, bf, (short)0, acc[0][tt], false, false);
      acc[1][tt] = __builtin_amdgcn_wmma_f32_16x16x32_bf16(
          false, af[1], false, bf, (short)0, acc[1][tt], false, false);
    }
  }

  float* Rb = R + b * DDIM * TDIM + t0 + lc;
#pragma unroll
  for (int mt = 0; mt < 2; ++mt)
#pragma unroll
    for (int tt = 0; tt < 2; ++tt)
#pragma unroll
      for (int r = 0; r < 8; ++r)
        Rb[(m0 + mt * 16 + hi * 8 + r) * TDIM + tt * 16] = acc[mt][tt][r];
}

// ---------------------------------------------------------------------------
extern "C" void kernel_launch(void* const* d_in, const int* in_sizes, int n_in,
                              void* d_out, int out_size, void* d_ws, size_t ws_size,
                              hipStream_t stream) {
  const float* Kp      = (const float*)d_in[0];
  const float* Vp      = (const float*)d_in[1];
  const float* Qp      = (const float*)d_in[2];
  const int*   force_p = (const int*)d_in[3];
  const int*   prevpos = (const int*)d_in[4];
  const float* prevA   = (const float*)d_in[5];
  const int*   ct_p    = (const int*)d_in[6];

  float* A   = (float*)d_out;
  float* R   = A + (size_t)BATCH * NDIM * TDIM;
  float* pos = R + (size_t)BATCH * DDIM * TDIM;

  attn_scores_softmax<<<dim3(TDIM / 64, BATCH), 1024, 0, stream>>>(
      Kp, Qp, prevA, force_p, ct_p, A);
  attn_finalize<<<BATCH, 256, 0, stream>>>(prevpos, force_p, ct_p, A, pos);
  r_gemm<<<dim3(TDIM / 32, DDIM / 256, BATCH), 256, 0, stream>>>(Vp, A, R);

  (void)in_sizes; (void)n_in; (void)out_size; (void)d_ws; (void)ws_size;
}
